// LocalAttention_48747878809985
// MI455X (gfx1250) — matrix operand order repaired and checked by reading, not compiled, not run on hardware
//
#include <hip/hip_runtime.h>
#include <math.h>

// ---------------------------------------------------------------------------
// Local (bucketed) attention forward for MI455X (gfx1250, wave32, WMMA).
//   q,k,v: [4,16,8192,64] f32.  bucket=128, window = [prev bucket, cur bucket]
//   out = softmax(mask(Q K^T * 64^-0.5)) V        (causal + pad mask)
// One workgroup (256 thr = 8 wave32) per (batch*head, bucket). K/V staged in
// LDS as f16; both GEMMs via v_wmma_f32_16x16x32_f16 (f32 accumulation).
// ---------------------------------------------------------------------------

typedef _Float16 h4   __attribute__((ext_vector_type(4)));
typedef _Float16 h8   __attribute__((ext_vector_type(8)));
typedef _Float16 v16h __attribute__((ext_vector_type(16)));
typedef float    v8f  __attribute__((ext_vector_type(8)));

#define BH_N   64      // 4 batches * 16 heads
#define SEQ    8192
#define HD     64      // head dim
#define BKTS   64      // 8192 / 128
#define BSZ    128     // bucket size
#define WIN    256     // look_backward=1, look_forward=0 -> 2 buckets
#define KSTR   72      // padded LDS stride for K  (f16 elems)
#define VSTR   264     // padded LDS stride for V^T (f16 elems)

static __device__ __forceinline__ v16h cat8(h8 lo, h8 hi) {
  return __builtin_shufflevector(lo, hi, 0,1,2,3,4,5,6,7,8,9,10,11,12,13,14,15);
}

__global__ __launch_bounds__(256)
void local_attn_fwd_48747878809985(const float* __restrict__ q,
                                   const float* __restrict__ k,
                                   const float* __restrict__ v,
                                   float* __restrict__ out) {
  __shared__ _Float16 kbuf[WIN * KSTR];      // [key][dim]   36,864 B
  __shared__ _Float16 vtbuf[HD * VSTR];      // [dim][key]   33,792 B
  __shared__ _Float16 pbuf[8 * 16 * 32];     // per-wave P bounce 8,192 B

  const int bh   = blockIdx.x / BKTS;
  const int bkt  = blockIdx.x - bh * BKTS;
  const int tid  = threadIdx.x;
  const int w    = tid >> 5;        // wave id 0..7 -> owns Q rows 16w..16w+15
  const int lane = tid & 31;
  const int h    = lane >> 4;       // half-wave (WMMA fragment half)
  const int ln   = lane & 15;

  // ---- stage K window [256][64] f32 -> f16 LDS, V window transposed --------
  {
    const size_t kvbase = ((size_t)bh * SEQ + (size_t)(bkt - 1) * BSZ) * HD;
    #pragma unroll
    for (int i = 0; i < 16; ++i) {
      const int cc   = tid + 256 * i;   // float4 chunk id (4096 total)
      const int key  = cc >> 4;         // 0..255 within window
      const int dim0 = (cc & 15) * 4;
      float4 kf = make_float4(0.f, 0.f, 0.f, 0.f);
      float4 vf = make_float4(0.f, 0.f, 0.f, 0.f);
      if (!((bkt == 0) && (key < BSZ))) {   // halo of bucket 0 is padding
        kf = *(const float4*)(k + kvbase + (size_t)key * HD + dim0);
        vf = *(const float4*)(v + kvbase + (size_t)key * HD + dim0);
      }
      h4 kh = { (_Float16)kf.x, (_Float16)kf.y, (_Float16)kf.z, (_Float16)kf.w };
      *(h4*)(&kbuf[key * KSTR + dim0]) = kh;
      vtbuf[(dim0 + 0) * VSTR + key] = (_Float16)vf.x;
      vtbuf[(dim0 + 1) * VSTR + key] = (_Float16)vf.y;
      vtbuf[(dim0 + 2) * VSTR + key] = (_Float16)vf.z;
      vtbuf[(dim0 + 3) * VSTR + key] = (_Float16)vf.w;
    }
  }
  __syncthreads();

  // ---- Q fragments straight from HBM (f32 -> f16, fold in e^-0.5) ----------
  // A-frag layout (16x32 f16): lanes0-15 m=ln hold k = {0..7,16..23};
  //                            lanes16-31 m=ln hold k = {8..15,24..31}.
  const float scale = 0.125f;   // 64^-0.5
  const float* qp = q + ((size_t)bh * SEQ + (size_t)bkt * BSZ + 16 * w + ln) * HD;
  v16h aq[2];
  #pragma unroll
  for (int c = 0; c < 2; ++c) {
    const int base = c * 32 + 8 * h;
    float4 f0 = *(const float4*)(qp + base);
    float4 f1 = *(const float4*)(qp + base + 4);
    float4 f2 = *(const float4*)(qp + base + 16);
    float4 f3 = *(const float4*)(qp + base + 20);
    v16h a;
    a[0]=(_Float16)(f0.x*scale); a[1]=(_Float16)(f0.y*scale);
    a[2]=(_Float16)(f0.z*scale); a[3]=(_Float16)(f0.w*scale);
    a[4]=(_Float16)(f1.x*scale); a[5]=(_Float16)(f1.y*scale);
    a[6]=(_Float16)(f1.z*scale); a[7]=(_Float16)(f1.w*scale);
    a[8]=(_Float16)(f2.x*scale); a[9]=(_Float16)(f2.y*scale);
    a[10]=(_Float16)(f2.z*scale); a[11]=(_Float16)(f2.w*scale);
    a[12]=(_Float16)(f3.x*scale); a[13]=(_Float16)(f3.y*scale);
    a[14]=(_Float16)(f3.z*scale); a[15]=(_Float16)(f3.w*scale);
    aq[c] = a;
  }

  // ---- S = Q K^T : 16 column tiles of 16 keys, K-dim 64 = 2 wmma steps -----
  // B-frag (32x16 f16): lane col n=ln, half h holds k = {8h..8h+7, 16+8h..16+8h+7}
  // from kbuf row (16j+ln) -> two b128 LDS loads per fragment.
  v8f sacc[16];
  #pragma unroll
  for (int j = 0; j < 16; ++j) {
    v8f acc = {0.f,0.f,0.f,0.f,0.f,0.f,0.f,0.f};
    #pragma unroll
    for (int c = 0; c < 2; ++c) {
      const _Float16* kp = &kbuf[(j * 16 + ln) * KSTR + c * 32 + 8 * h];
      v16h b = cat8(*(const h8*)kp, *(const h8*)(kp + 16));
      acc = __builtin_amdgcn_wmma_f32_16x16x32_f16(false, aq[c], false, b,
                                                   (short)0, acc, false, false);
    }
    sacc[j] = acc;
  }

  // ---- mask + softmax (rows striped over 16 lanes in C-layout) -------------
  // C/D layout: element r of v8f = row m = r + 8h, col n = 16j + ln.
  const float NEGMAX = -3.402823466e+38f;
  float rmax[8], rsum[8], rinv[8];
  #pragma unroll
  for (int r = 0; r < 8; ++r) rmax[r] = NEGMAX;
  #pragma unroll
  for (int j = 0; j < 16; ++j) {
    const int col = j * 16 + ln;            // window position 0..255
    #pragma unroll
    for (int r = 0; r < 8; ++r) {
      const int qi = 16 * w + r + 8 * h;    // query row 0..127
      const bool masked = ((bkt == 0) && (col < BSZ)) || (col > qi + BSZ);
      const float s = masked ? NEGMAX : sacc[j][r];
      sacc[j][r] = s;
      rmax[r] = fmaxf(rmax[r], s);
    }
  }
  #pragma unroll
  for (int r = 0; r < 8; ++r) {
    #pragma unroll
    for (int m = 1; m < 16; m <<= 1)
      rmax[r] = fmaxf(rmax[r], __shfl_xor(rmax[r], m, 16));
    rsum[r] = 0.f;
  }
  #pragma unroll
  for (int j = 0; j < 16; ++j) {
    #pragma unroll
    for (int r = 0; r < 8; ++r) {
      const float p = __expf(sacc[j][r] - rmax[r]);
      sacc[j][r] = p;
      rsum[r] += p;
    }
  }
  #pragma unroll
  for (int r = 0; r < 8; ++r) {
    #pragma unroll
    for (int m = 1; m < 16; m <<= 1)
      rsum[r] += __shfl_xor(rsum[r], m, 16);
    rinv[r] = 1.f / rsum[r];
  }

  // ---- O = P V : stream P through a wave-private LDS bounce ----------------
  v8f oacc[4];
  #pragma unroll
  for (int t = 0; t < 4; ++t) oacc[t] = (v8f){0.f,0.f,0.f,0.f,0.f,0.f,0.f,0.f};

  const int pwb = w * 512;                  // 16 rows x 32 cols per wave
  #pragma unroll
  for (int c = 0; c < 8; ++c) {             // key chunks of 32
    #pragma unroll
    for (int jl = 0; jl < 2; ++jl) {        // score tiles 2c, 2c+1
      const int j = 2 * c + jl;
      #pragma unroll
      for (int r = 0; r < 8; ++r)
        pbuf[pwb + (r + 8 * h) * 32 + jl * 16 + ln] = (_Float16)sacc[j][r];
    }
    const _Float16* pp = &pbuf[pwb + ln * 32 + 8 * h];
    v16h a = cat8(*(const h8*)pp, *(const h8*)(pp + 16));
    #pragma unroll
    for (int t = 0; t < 4; ++t) {           // 4 output dim tiles of 16
      const _Float16* vp = &vtbuf[(t * 16 + ln) * VSTR + c * 32 + 8 * h];
      v16h b = cat8(*(const h8*)vp, *(const h8*)(vp + 16));
      oacc[t] = __builtin_amdgcn_wmma_f32_16x16x32_f16(false, a, false, b,
                                                       (short)0, oacc[t],
                                                       false, false);
    }
  }

  // ---- scale by 1/rowsum (already in C-layout) and store -------------------
  #pragma unroll
  for (int t = 0; t < 4; ++t) {
    #pragma unroll
    for (int r = 0; r < 8; ++r) {
      const int qi = 16 * w + r + 8 * h;
      out[((size_t)bh * SEQ + (size_t)bkt * BSZ + qi) * HD + t * 16 + ln] =
          oacc[t][r] * rinv[r];
    }
  }
}

extern "C" void kernel_launch(void* const* d_in, const int* in_sizes, int n_in,
                              void* d_out, int out_size, void* d_ws, size_t ws_size,
                              hipStream_t stream) {
  (void)in_sizes; (void)n_in; (void)out_size; (void)d_ws; (void)ws_size;
  const float* q = (const float*)d_in[0];
  const float* k = (const float*)d_in[1];
  const float* v = (const float*)d_in[2];
  float* out = (float*)d_out;
  dim3 grid(BH_N * BKTS);   // 4096 (bh, bucket) tiles
  dim3 block(256);          // 8 wave32
  local_attn_fwd_48747878809985<<<grid, block, 0, stream>>>(q, k, v, out);
}
